// DynamicRankMixtralSparseMoeBlockWrapper_87668872446487
// MI455X (gfx1250) — matrix operand order repaired and checked by reading, not compiled, not run on hardware
//
#include <hip/hip_runtime.h>
#include <hip/hip_bf16.h>

// ---------------- problem constants ----------------
#define T_TOK   2048      // B*S tokens
#define HDIM    1024
#define FDIM    3584
#define NEXP    8
#define NHEAD   16
#define SEQ     2048
#define NLOW    204       // int(0.1 * 2048)
#define MAXROWS 4352      // <= 4096 selected rows + 8 experts * 15 pad, 128-aligned

// ---------------- CDNA5 WMMA types ----------------
typedef __attribute__((ext_vector_type(16))) __bf16 v16bf;
typedef __attribute__((ext_vector_type(8)))  float  v8f;

union BF16x16 { v16bf v; unsigned short h[16]; unsigned int u[8]; };

__device__ __forceinline__ unsigned short f2bf(float f) {
  unsigned int u = __float_as_uint(f);
  u += 0x7FFFu + ((u >> 16) & 1u);       // round-to-nearest-even
  return (unsigned short)(u >> 16);
}
__device__ __forceinline__ unsigned int pack2bf(float a, float b) {
  return (unsigned int)f2bf(a) | ((unsigned int)f2bf(b) << 16);
}

// ---------------- kernel 0: zero the dense output region ----------------
__global__ __launch_bounds__(256) void zero_out_kernel(float* __restrict__ out, int n) {
  int i = blockIdx.x * 256 + threadIdx.x;
  if (i < n) out[i] = 0.0f;
}

// ---------------- kernel 0b: one-time f32 -> bf16 stream (weights + x) ----------------
// halves all weight bandwidth for the GEMM passes and removes every pack op
// from the WMMA K-loops.
__global__ __launch_bounds__(256) void cvt_bf16_kernel(const float* __restrict__ src,
                                                       unsigned short* __restrict__ dst,
                                                       int n4) {
  int i = blockIdx.x * 256 + threadIdx.x;
  if (i < n4) {
    float4 f = ((const float4*)src)[i];
    uint2 o;
    o.x = pack2bf(f.x, f.y);
    o.y = pack2bf(f.z, f.w);
    ((uint2*)dst)[i] = o;
  }
}

// ---------------- kernel 1: router GEMV + softmax + top-2 + beta prune ----------------
// one wave32 per token
__global__ __launch_bounds__(128) void router_kernel(
    const float* __restrict__ x, const float* __restrict__ gw,
    float* __restrict__ logits_out, int* __restrict__ sel0, int* __restrict__ sel1,
    float* __restrict__ rw0, float* __restrict__ rw1, int* __restrict__ flag) {
  const int t = blockIdx.x * 4 + (threadIdx.x >> 5);
  const int lane = threadIdx.x & 31;
  const float* xr = x + (size_t)t * HDIM;
  float lg[NEXP];
#pragma unroll
  for (int e = 0; e < NEXP; ++e) {
    const float* g = gw + e * HDIM;
    float s = 0.f;
    for (int i = lane; i < HDIM; i += 32) s = fmaf(xr[i], g[i], s);
#pragma unroll
    for (int o = 16; o > 0; o >>= 1) s += __shfl_xor(s, o, 32);
    lg[e] = s;
  }
  float mx = lg[0];
#pragma unroll
  for (int e = 1; e < NEXP; ++e) mx = fmaxf(mx, lg[e]);
  float p[NEXP]; float se = 0.f;
#pragma unroll
  for (int e = 0; e < NEXP; ++e) { p[e] = __expf(lg[e] - mx); se += p[e]; }
  const float inv = 1.f / se;
  int i0 = 0; float p0 = p[0];
#pragma unroll
  for (int e = 1; e < NEXP; ++e) if (p[e] > p0) { p0 = p[e]; i0 = e; }
  int i1 = (i0 == 0) ? 1 : 0; float p1v = p[i1];
#pragma unroll
  for (int e = 0; e < NEXP; ++e) if (e != i0 && p[e] > p1v) { p1v = p[e]; i1 = e; }
  p0 *= inv; p1v *= inv;
  if (lane < NEXP) logits_out[(size_t)t * NEXP + lane] = lg[lane];
  if (lane == 0) {
    sel0[t] = i0; sel1[t] = i1; rw0[t] = p0; rw1[t] = p1v;
    flag[t] = (p1v < 0.5f * p0) ? 0 : 1;           // BETA = 0.5
  }
}

// ---------------- kernel 2: attention token scores (streams 256 MB) ----------------
__global__ __launch_bounds__(256) void attn_score_kernel(
    const float* __restrict__ attn, const float* __restrict__ x,
    float* __restrict__ scores) {
  const int i = blockIdx.x;
  const int tid = threadIdx.x;
  float s = 0.f;
  for (int h = 0; h < NHEAD; ++h) {
    const float4* row = (const float4*)(attn + ((size_t)h * SEQ + i) * SEQ);
    for (int j = tid; j < SEQ / 4; j += 256) {
      float4 v = row[j];
      s += (v.x + v.y) + (v.z + v.w);
    }
  }
  float l1 = 0.f;
  const float* xr = x + (size_t)i * HDIM;
  for (int j = tid; j < HDIM; j += 256) l1 += fabsf(xr[j]);

  __shared__ float red[256];
  red[tid] = s; __syncthreads();
  for (int o = 128; o > 0; o >>= 1) { if (tid < o) red[tid] += red[tid + o]; __syncthreads(); }
  const float asum = red[0]; __syncthreads();
  red[tid] = l1; __syncthreads();
  for (int o = 128; o > 0; o >>= 1) { if (tid < o) red[tid] += red[tid + o]; __syncthreads(); }
  if (tid == 0) scores[i] = (asum * (1.0f / 16.0f) / (float)(SEQ - i)) * red[0];
}

// ---------------- kernel 3: threshold select + routing finalize + expert lists ----------------
// single block, 1024 threads; bitonic sort of 2048 scores in LDS
__global__ __launch_bounds__(1024) void select_kernel(
    const float* __restrict__ scores, const int* __restrict__ sel0,
    const int* __restrict__ sel1, const float* __restrict__ rw0,
    const float* __restrict__ rw1, const int* __restrict__ flag,
    int* __restrict__ hdr, int* __restrict__ row_tok, float* __restrict__ row_wt) {
  __shared__ float srt[T_TOK];
  __shared__ int cnt[NEXP], cnt2[NEXP], off[NEXP + 1];
  const int tid = threadIdx.x;
  srt[tid] = scores[tid];
  srt[tid + 1024] = scores[tid + 1024];
  if (tid < NEXP) { cnt[tid] = 0; cnt2[tid] = 0; }
  __syncthreads();

  for (int k = 2; k <= T_TOK; k <<= 1) {
    for (int j = k >> 1; j > 0; j >>= 1) {
#pragma unroll
      for (int b = 0; b < 2; ++b) {
        int i = tid + b * 1024;
        int ixj = i ^ j;
        if (ixj > i) {
          bool up = ((i & k) == 0);
          float a = srt[i], c = srt[ixj];
          if ((a > c) == up) { srt[i] = c; srt[ixj] = a; }
        }
      }
      __syncthreads();
    }
  }
  const float low_thr  = srt[NLOW - 1];         // 204th smallest
  const float high_thr = srt[T_TOK - NLOW];     // 204th largest

  int   mt[2], ms0[2], ms1[2];
  float mc0[2], mc1[2];
#pragma unroll
  for (int q = 0; q < 2; ++q) {
    int t = tid + q * 1024;
    float s = scores[t];
    float r0 = 1.f;
    float r1 = flag[t] ? 1.f : 0.f;
    if (s <= low_thr)  { r0 = 0.f; r1 = 0.f; }  // mask_all
    if (s >= high_thr) { r0 = 1.f; r1 = 1.f; }  // keep_all overrides
    float a = rw0[t] * r0, b = rw1[t] * r1;
    float d = a + b;
    float c0 = 0.f, c1 = 0.f;
    if (d > 0.f) { c0 = a / d; c1 = b / d; }
    mt[q] = t; mc0[q] = c0; mc1[q] = c1; ms0[q] = sel0[t]; ms1[q] = sel1[t];
    if (c0 > 0.f) atomicAdd(&cnt[ms0[q]], 1);
    if (c1 > 0.f) atomicAdd(&cnt[ms1[q]], 1);
  }
  __syncthreads();
  if (tid == 0) {
    off[0] = 0;
    for (int e = 0; e < NEXP; ++e) off[e + 1] = off[e] + ((cnt[e] + 15) & ~15);
    for (int e = 0; e < NEXP; ++e) hdr[e] = cnt[e];
    for (int e = 0; e <= NEXP; ++e) hdr[8 + e] = off[e];
    hdr[17] = off[NEXP];                        // total 16-aligned rows
  }
  __syncthreads();
#pragma unroll
  for (int q = 0; q < 2; ++q) {
    if (mc0[q] > 0.f) {
      int p = atomicAdd(&cnt2[ms0[q]], 1);
      int r = off[ms0[q]] + p;
      row_tok[r] = mt[q]; row_wt[r] = mc0[q];
    }
    if (mc1[q] > 0.f) {
      int p = atomicAdd(&cnt2[ms1[q]], 1);
      int r = off[ms1[q]] + p;
      row_tok[r] = mt[q]; row_wt[r] = mc1[q];
    }
  }
  __syncthreads();
  for (int e = 0; e < NEXP; ++e)
    for (int r = off[e] + cnt[e] + tid; r < off[e + 1]; r += 1024) {
      row_tok[r] = 0; row_wt[r] = 0.f;          // zero-weight padding rows
    }
}

// ---------------- kernel 4: up-proj + SwiGLU (two WMMA per K-step) ----------------
// block: 256 thr = 8 waves; tile M=128 gathered rows x N=16 F-cols; K = HDIM
// all operands pre-converted bf16 -> K-loop is pure b128 loads + WMMA, no VALU packing
__global__ __launch_bounds__(256) void expert_up_kernel(
    const unsigned short* __restrict__ xb, const unsigned short* __restrict__ w1b,
    const unsigned short* __restrict__ w3b, const int* __restrict__ hdr,
    const int* __restrict__ row_tok, unsigned short* __restrict__ Aws) {
  const int total = hdr[17];
  const int rblk = blockIdx.y * 128;
  if (rblk >= total) return;
  const int tid = threadIdx.x;
  const int wave = tid >> 5, lane = tid & 31;
  const int half = lane >> 4, nn = lane & 15;
  const int rsub = rblk + wave * 16;
  const int ncol = blockIdx.x * 16 + nn;

  const int rrow = rsub + nn;                       // this lane's gathered A row
  const int tok = (rrow < total) ? row_tok[rrow] : 0;

  int rq = rsub < total ? rsub : (total - 1);
  int e = 0;
#pragma unroll
  for (int i = 1; i < NEXP; ++i) if (rq >= hdr[8 + i]) e = i;

  const unsigned short* xrow  = xb  + (size_t)tok * HDIM;
  const unsigned short* w1row = w1b + ((size_t)e * FDIM + ncol) * HDIM;
  const unsigned short* w3row = w3b + ((size_t)e * FDIM + ncol) * HDIM;

  v8f acc_h = {}; v8f acc_g = {};
  for (int kk = 0; kk < HDIM; kk += 32) {
    BF16x16 af;                                     // A 16x32 bf16 fragment: 2x b128
    uint4 a0 = *(const uint4*)(xrow + kk + (half << 4));
    uint4 a1 = *(const uint4*)(xrow + kk + 8 + (half << 4));
    af.u[0] = a0.x; af.u[1] = a0.y; af.u[2] = a0.z; af.u[3] = a0.w;
    af.u[4] = a1.x; af.u[5] = a1.y; af.u[6] = a1.z; af.u[7] = a1.w;

    const uint4* p1 = (const uint4*)(w1row + kk + (half << 4));
    const uint4* p3 = (const uint4*)(w3row + kk + (half << 4));
    __builtin_prefetch(w1row + kk + 256, 0, 0);
    __builtin_prefetch(w3row + kk + 256, 0, 0);
    BF16x16 b1, b3;                                 // B 32x16 bf16: 2x b128 each
    uint4 q0 = p1[0], q1 = p1[1], r0 = p3[0], r1 = p3[1];
    b1.u[0] = q0.x; b1.u[1] = q0.y; b1.u[2] = q0.z; b1.u[3] = q0.w;
    b1.u[4] = q1.x; b1.u[5] = q1.y; b1.u[6] = q1.z; b1.u[7] = q1.w;
    b3.u[0] = r0.x; b3.u[1] = r0.y; b3.u[2] = r0.z; b3.u[3] = r0.w;
    b3.u[4] = r1.x; b3.u[5] = r1.y; b3.u[6] = r1.z; b3.u[7] = r1.w;

    acc_h = __builtin_amdgcn_wmma_f32_16x16x32_bf16(false, af.v, false, b1.v,
                                                    (short)0, acc_h, false, false);
    acc_g = __builtin_amdgcn_wmma_f32_16x16x32_bf16(false, af.v, false, b3.v,
                                                    (short)0, acc_g, false, false);
  }

  if (rsub < total) {
#pragma unroll
    for (int r = 0; r < 8; ++r) {                   // silu(h)*g via fast v_rcp_f32
      float hv = acc_h[r], gv = acc_g[r];
      float sig = __builtin_amdgcn_rcpf(1.0f + __expf(-hv));
      float a = hv * sig * gv;
      int m = (half << 3) + r;
      Aws[(size_t)(rsub + m) * FDIM + ncol] = f2bf(a);
    }
  }
}

// ---------------- kernel 5: down-proj + weighted scatter ----------------
// block: 256 thr = 8 waves; tile M=128 rows x N=16 H-cols; K = FDIM
__global__ __launch_bounds__(256) void expert_down_kernel(
    const unsigned short* __restrict__ Aws, const unsigned short* __restrict__ w2b,
    const int* __restrict__ hdr, const int* __restrict__ row_tok,
    const float* __restrict__ row_wt, float* __restrict__ out) {
  const int total = hdr[17];
  const int rblk = blockIdx.y * 128;
  if (rblk >= total) return;
  const int tid = threadIdx.x;
  const int wave = tid >> 5, lane = tid & 31;
  const int half = lane >> 4, nn = lane & 15;
  const int rsub = rblk + wave * 16;
  const int ncol = blockIdx.x * 16 + nn;

  __shared__ int toks[128];
  __shared__ float wts[128];
  if (tid < 128) {
    int r = rblk + tid;
    toks[tid] = (r < total) ? row_tok[r] : 0;
    wts[tid]  = (r < total) ? row_wt[r] : 0.f;
  }
  __syncthreads();

  int rq = rsub < total ? rsub : (total - 1);
  int e = 0;
#pragma unroll
  for (int i = 1; i < NEXP; ++i) if (rq >= hdr[8 + i]) e = i;

  const unsigned short* w2row = w2b + ((size_t)e * HDIM + ncol) * FDIM;
  const unsigned short* arow  = Aws + (size_t)(rsub + nn) * FDIM;

  v8f acc = {};
  for (int kk = 0; kk < FDIM; kk += 32) {
    BF16x16 af;                                     // bf16 activations: 2x b128
    uint4 a0 = *(const uint4*)(arow + kk + (half << 4));
    uint4 a1 = *(const uint4*)(arow + kk + 8 + (half << 4));
    af.u[0] = a0.x; af.u[1] = a0.y; af.u[2] = a0.z; af.u[3] = a0.w;
    af.u[4] = a1.x; af.u[5] = a1.y; af.u[6] = a1.z; af.u[7] = a1.w;

    const uint4* p2 = (const uint4*)(w2row + kk + (half << 4));
    __builtin_prefetch(w2row + kk + 256, 0, 0);
    BF16x16 b2;
    uint4 q0 = p2[0], q1 = p2[1];
    b2.u[0] = q0.x; b2.u[1] = q0.y; b2.u[2] = q0.z; b2.u[3] = q0.w;
    b2.u[4] = q1.x; b2.u[5] = q1.y; b2.u[6] = q1.z; b2.u[7] = q1.w;

    acc = __builtin_amdgcn_wmma_f32_16x16x32_bf16(false, af.v, false, b2.v,
                                                  (short)0, acc, false, false);
  }

  if (rsub < total) {
#pragma unroll
    for (int r = 0; r < 8; ++r) {
      int m = (wave << 4) + (half << 3) + r;
      float v = wts[m] * acc[r];                    // <=2 contributions/elem, commutative
      atomicAdd(&out[(size_t)toks[m] * HDIM + ncol], v);
    }
  }
}

// ---------------- host-side launch ----------------
extern "C" void kernel_launch(void* const* d_in, const int* in_sizes, int n_in,
                              void* d_out, int out_size, void* d_ws, size_t ws_size,
                              hipStream_t stream) {
  (void)in_sizes; (void)n_in; (void)out_size; (void)ws_size;
  const float* x    = (const float*)d_in[0];   // [1,2048,1024]
  const float* attn = (const float*)d_in[1];   // [1,16,2048,2048]
  const float* gw   = (const float*)d_in[2];   // [8,1024]
  const float* w1   = (const float*)d_in[3];   // [8,3584,1024]
  const float* w2   = (const float*)d_in[4];   // [8,1024,3584]
  const float* w3   = (const float*)d_in[5];   // [8,3584,1024]

  float* out    = (float*)d_out;
  float* logits = out + (size_t)T_TOK * HDIM;  // tuple: (out, router_logits)

  char* ws = (char*)d_ws;
  float*          scores  = (float*)(ws + 0);        // 2048 f32
  int*            sel0    = (int*)  (ws + 8192);
  int*            sel1    = (int*)  (ws + 16384);
  float*          rw0     = (float*)(ws + 24576);
  float*          rw1     = (float*)(ws + 32768);
  int*            flag    = (int*)  (ws + 40960);
  int*            hdr     = (int*)  (ws + 49152);    // cnt[8], off[9], total
  int*            row_tok = (int*)  (ws + 49408);    // MAXROWS
  float*          row_wt  = (float*)(ws + 66816);    // MAXROWS
  unsigned short* Aws     = (unsigned short*)(ws + 84480);     // MAXROWS*FDIM bf16
  unsigned short* w1b     = (unsigned short*)(ws + 31279616);  // 8*3584*1024 bf16
  unsigned short* w3b     = (unsigned short*)(ws + 89999872);
  unsigned short* w2b     = (unsigned short*)(ws + 148720128);
  unsigned short* xb      = (unsigned short*)(ws + 207440384); // 2048*1024 bf16

  const int nw4 = (NEXP * FDIM * HDIM) / 4;          // 7,340,032 float4 per weight
  const int nx4 = (T_TOK * HDIM) / 4;                // 524,288 float4 for x

  zero_out_kernel<<<(T_TOK * HDIM + 255) / 256, 256, 0, stream>>>(out, T_TOK * HDIM);
  cvt_bf16_kernel<<<(nw4 + 255) / 256, 256, 0, stream>>>(w1, w1b, nw4);
  cvt_bf16_kernel<<<(nw4 + 255) / 256, 256, 0, stream>>>(w3, w3b, nw4);
  cvt_bf16_kernel<<<(nw4 + 255) / 256, 256, 0, stream>>>(w2, w2b, nw4);
  cvt_bf16_kernel<<<(nx4 + 255) / 256, 256, 0, stream>>>(x, xb, nx4);
  router_kernel<<<T_TOK / 4, 128, 0, stream>>>(x, gw, logits, sel0, sel1, rw0, rw1, flag);
  attn_score_kernel<<<SEQ, 256, 0, stream>>>(attn, x, scores);
  select_kernel<<<1, 1024, 0, stream>>>(scores, sel0, sel1, rw0, rw1, flag,
                                        hdr, row_tok, row_wt);
  dim3 ga(FDIM / 16, MAXROWS / 128);   // 224 x 34
  expert_up_kernel<<<ga, 256, 0, stream>>>(xb, w1b, w3b, hdr, row_tok, Aws);
  dim3 gb(HDIM / 16, MAXROWS / 128);   // 64 x 34
  expert_down_kernel<<<gb, 256, 0, stream>>>(Aws, w2b, hdr, row_tok, row_wt, out);
}
